// MultiHeadAttention_2851858284900
// MI455X (gfx1250) — matrix operand order repaired and checked by reading, not compiled, not run on hardware
//
#include <hip/hip_runtime.h>
#include <hip/hip_bf16.h>
#include <math.h>
#include <stdint.h>

#define EMBED   2048
#define D_COMPR 512
#define D_ROPE  128
#define NHEADS  8
#define HDIM    64
#define BATCH   4
#define SEQ     2048
#define NTOK    (BATCH * SEQ)

typedef __bf16 bf16;
typedef __attribute__((ext_vector_type(16))) __bf16 v16bf;
typedef __attribute__((ext_vector_type(8)))  float  v8f;
typedef __attribute__((ext_vector_type(4)))  unsigned int v4u;
typedef __attribute__((ext_vector_type(4)))  int v4i;
typedef __attribute__((ext_vector_type(8)))  int v8i;

#if defined(__has_builtin)
#  if __has_builtin(__builtin_amdgcn_tensor_load_to_lds)
#    define HAVE_TDM 1
#  endif
#endif

// ---------------------------------------------------------------------------
// helpers
// ---------------------------------------------------------------------------
__device__ __forceinline__ float gelu_exact(float x) {
  return 0.5f * x * (1.0f + erff(x * 0.70710678118654752440f));
}

__device__ __forceinline__ v8f wmma_bf16(v16bf a, v16bf b, v8f c) {
  return __builtin_amdgcn_wmma_f32_16x16x32_bf16(
      /*neg_a=*/false, a, /*neg_b=*/false, b,
      /*c_mod=*/(short)0, c, /*reuse_a=*/false, /*reuse_b=*/false);
}

// A-matrix 16x32 bf16 fragment from row-major src (row stride ld).
// element j -> row (lane&15), col j + (lane>=16?16:0)
__device__ __forceinline__ v16bf load_a_frag(const bf16* __restrict__ src,
                                             int ld, int lane) {
  v16bf f;
  const bf16* row = src + (size_t)(lane & 15) * ld + ((lane >> 4) ? 16 : 0);
#pragma unroll
  for (int j = 0; j < 16; ++j) {
    int k = j;
    f[j] = row[k];
  }
  return f;
}

// B-matrix 32x16 fragment where B[k][n] = w[n*ld + k]  (B = W^T, W row-major
// [N,K]).  element j -> k = j + (lane>=16?16:0), n = lane&15.
__device__ __forceinline__ v16bf load_bT_frag(const bf16* __restrict__ w,
                                              int ld, int lane) {
  v16bf f;
  const bf16* row = w + (size_t)(lane & 15) * ld + ((lane >> 4) ? 16 : 0);
#pragma unroll
  for (int j = 0; j < 16; ++j) f[j] = row[j];
  return f;
}

// B-matrix 32x16 fragment from row-major K x N source (B[k][n] = v[k*ld + n]).
__device__ __forceinline__ v16bf load_b_frag(const bf16* __restrict__ v,
                                             int ld, int lane) {
  v16bf f;
  const bf16* col = v + (size_t)((lane >> 4) ? 16 : 0) * ld + (lane & 15);
#pragma unroll
  for (int j = 0; j < 16; ++j) f[j] = col[(size_t)j * ld];
  return f;
}

// Low 32 bits of a generic LDS pointer are the LDS byte offset (ISA 10.2).
__device__ __forceinline__ unsigned lds_byte_off(const void* p) {
  return (unsigned)(uintptr_t)p;
}

// ---- CDNA5 Tensor Data Mover: one descriptor moves a 2-D bf16 tile --------
#if HAVE_TDM
// Load a (rows x 64) bf16 tile whose rows are D_COMPR apart into LDS
// (packed rows of 64 bf16).  D# layout per CDNA5 ISA.
__device__ __forceinline__ void tdm_load_tile(unsigned lds_off,
                                              const void* gaddr,
                                              unsigned rows) {
  const unsigned long long ga = (unsigned long long)(uintptr_t)gaddr;
  v4u g0;
  g0[0] = 1u;                                   // count=1, user descriptor
  g0[1] = lds_off;                              // lds_addr (bytes)
  g0[2] = (unsigned)(ga & 0xFFFFFFFFu);         // global_addr[31:0]
  g0[3] = (unsigned)((ga >> 32) & 0x01FFFFFFu)  // global_addr[56:32]
          | (2u << 30);                         // type = 2 ("image")
  v8i g1;
  g1[0] = (int)(1u << 16);          // workgroup_mask=0, data_size=1 (2 bytes)
  g1[1] = (int)(64u << 16);         // tensor_dim0[15:0]=64 in bits[31:16]
  g1[2] = (int)(rows << 16);        // tensor_dim0[31:16]=0 | tensor_dim1[15:0]
  g1[3] = (int)(64u << 16);         // tensor_dim1[31:16]=0 | tile_dim0=64
  g1[4] = (int)rows;                // tile_dim1=rows, tile_dim2=0
  g1[5] = (int)D_COMPR;             // tensor_dim0_stride[31:0] = 512
  g1[6] = 0;                        // stride0[47:32]=0, stride1[15:0]=0
  g1[7] = 0;                        // tensor_dim1_stride[47:16]=0
  v4i gz = {0, 0, 0, 0};            // groups 2/3: unused dims for 2-D tile
#if __has_include(<hip/amd_detail/amd_gfx1250_TDM.h>)
  v8i gz8 = {0, 0, 0, 0, 0, 0, 0, 0};
  __builtin_amdgcn_tensor_load_to_lds(g0, g1, gz, gz, gz8, 0);
#else
  __builtin_amdgcn_tensor_load_to_lds(g0, g1, gz, gz, 0);
#endif
}
#define WAIT_STAGE_PREFETCH() __builtin_amdgcn_s_wait_tensorcnt((short)2)
#define WAIT_STAGE_ALL()      __builtin_amdgcn_s_wait_tensorcnt((short)0)
#else
// Fallback: per-lane async global->LDS copies (ASYNCcnt-tracked).
__device__ __forceinline__ void async_copy_b128(unsigned lds_off,
                                                const void* gaddr) {
  asm volatile("global_load_async_to_lds_b128 %0, %1, off"
               :
               : "v"(lds_off), "v"((unsigned long long)(uintptr_t)gaddr)
               : "memory");
}
#define WAIT_STAGE_PREFETCH() asm volatile("s_wait_asynccnt 0x10" ::: "memory")
#define WAIT_STAGE_ALL()      asm volatile("s_wait_asynccnt 0x0" ::: "memory")
#endif

// ---------------------------------------------------------------------------
// fp32 -> bf16 conversion
// ---------------------------------------------------------------------------
__global__ void f32_to_bf16_kernel(const float* __restrict__ src,
                                   bf16* __restrict__ dst, long n) {
  long i = (long)blockIdx.x * blockDim.x + threadIdx.x;
  if (i < n) dst[i] = (bf16)src[i];
}

// ---------------------------------------------------------------------------
// Generic bf16 WMMA GEMM:  C[m][n] = sum_k A[m][k] * W[n][k] + bias[n]
// wave = 32x32 tile, block = 4 waves stacked in M (128x32).
// ---------------------------------------------------------------------------
__global__ void __launch_bounds__(128)
gemm_bf16_wmma(const bf16* __restrict__ A, int lda,
               const bf16* __restrict__ W, int ldw,
               const float* __restrict__ bias,
               bf16* __restrict__ outb, float* __restrict__ outf,
               int ldc, int coloff, int Kdim, int do_gelu) {
  const int lane = threadIdx.x & 31;
  const int wave = threadIdx.x >> 5;
  const int m0 = blockIdx.x * 128 + wave * 32;
  const int n0 = blockIdx.y * 32;

  v8f acc[2][2] = {};

  for (int k0 = 0; k0 < Kdim; k0 += 32) {
    v16bf a0 = load_a_frag(A + (size_t)m0 * lda + k0, lda, lane);
    v16bf a1 = load_a_frag(A + (size_t)(m0 + 16) * lda + k0, lda, lane);
    v16bf b0 = load_bT_frag(W + (size_t)n0 * ldw + k0, ldw, lane);
    v16bf b1 = load_bT_frag(W + (size_t)(n0 + 16) * ldw + k0, ldw, lane);
    acc[0][0] = wmma_bf16(a0, b0, acc[0][0]);
    acc[0][1] = wmma_bf16(a0, b1, acc[0][1]);
    acc[1][0] = wmma_bf16(a1, b0, acc[1][0]);
    acc[1][1] = wmma_bf16(a1, b1, acc[1][1]);
  }

  const int half = lane >> 4;
  const int nl   = lane & 15;
#pragma unroll
  for (int i = 0; i < 2; ++i) {
#pragma unroll
    for (int j = 0; j < 2; ++j) {
      const int colg = n0 + j * 16 + nl;
      const float bv = bias ? bias[colg] : 0.0f;
#pragma unroll
      for (int r = 0; r < 8; ++r) {
        const int rowg = m0 + i * 16 + r + half * 8;
        float v = acc[i][j][r] + bv;
        if (do_gelu) v = gelu_exact(v);
        if (outb) outb[(size_t)rowg * ldc + coloff + colg] = (bf16)v;
        else      outf[(size_t)rowg * ldc + coloff + colg] = v;
      }
    }
  }
}

// ---------------------------------------------------------------------------
// RoPE + gelu, writing bf16 into columns [384..511] of the Q/K concat buffer.
// ---------------------------------------------------------------------------
__global__ void rope_gelu_kernel(const float* __restrict__ src,
                                 bf16* __restrict__ dst) {
  int idx = blockIdx.x * blockDim.x + threadIdx.x;  // token*64 + j
  int token = idx >> 6;
  int j = idx & 63;
  if (token >= NTOK) return;
  int pos = token & (SEQ - 1);  // token = b*SEQ + s
  float invf = __powf(10000.0f, -(float)(2 * j) * (1.0f / (float)D_ROPE));
  float f = (float)pos * invf;
  float c, s;
  __sincosf(f, &s, &c);
  float x1 = src[(size_t)token * D_ROPE + j];
  float x2 = src[(size_t)token * D_ROPE + j + 64];
  float o1 = gelu_exact(x1 * c - x2 * s);
  float o2 = gelu_exact(x2 * c + x1 * s);
  dst[(size_t)token * D_COMPR + (D_COMPR - D_ROPE) + j]      = (bf16)o1;
  dst[(size_t)token * D_COMPR + (D_COMPR - D_ROPE) + 64 + j] = (bf16)o2;
}

// ---------------------------------------------------------------------------
// Flash-style causal attention, one wave per 16-row q-tile per (b,h).
// K and V tiles (32x64 bf16, 4KB each) are staged into LDS, double-buffered:
// with TDM, one tensor_load_to_lds descriptor per tile (2 per chunk) and
// s_wait_tensorcnt 2 keeps only the next chunk's DMA in flight while this
// chunk runs its 8 WMMAs + online softmax.
// ---------------------------------------------------------------------------
__global__ void __launch_bounds__(32)
mla_attention_kernel(const bf16* __restrict__ Q, const bf16* __restrict__ Km,
                     const bf16* __restrict__ V, bf16* __restrict__ O) {
  const int lane = threadIdx.x & 31;
  const int q0 = blockIdx.x * 16;
  const int h = blockIdx.y;
  const int b = blockIdx.z;
  const size_t base = (size_t)b * SEQ * D_COMPR + (size_t)h * HDIM;
  const bf16* Qh = Q + base;
  const bf16* Kh = Km + base;
  const bf16* Vh = V + base;

  __shared__ bf16 Klds[2][32 * 64];
  __shared__ bf16 Vlds[2][32 * 64];
  __shared__ bf16 Plds[16 * 32];

  // issue staging of K/V rows [k0, k0+32) into buffer `buf`
  auto stage = [&](int k0, int buf) {
#if HAVE_TDM
    tdm_load_tile(lds_byte_off(&Klds[buf][0]),
                  Kh + (size_t)k0 * D_COMPR, 32u);
    tdm_load_tile(lds_byte_off(&Vlds[buf][0]),
                  Vh + (size_t)k0 * D_COMPR, 32u);
#else
    const unsigned kbase = lds_byte_off(&Klds[buf][0]);
    const unsigned vbase = lds_byte_off(&Vlds[buf][0]);
#pragma unroll
    for (int i = 0; i < 8; ++i) {
      const int seg = lane + 32 * i;  // 0..255, 16B each (32 rows x 8)
      const int row = seg >> 3;
      const int sc  = seg & 7;
      const unsigned loff = (unsigned)(row * 128 + sc * 16);
      async_copy_b128(kbase + loff, Kh + (size_t)(k0 + row) * D_COMPR + sc * 8);
      async_copy_b128(vbase + loff, Vh + (size_t)(k0 + row) * D_COMPR + sc * 8);
    }
#endif
  };

  // Q fragments for the whole tile: d-chunks [0,32) and [32,64)
  v16bf aQ0 = load_a_frag(Qh + (size_t)q0 * D_COMPR + 0, D_COMPR, lane);
  v16bf aQ1 = load_a_frag(Qh + (size_t)q0 * D_COMPR + 32, D_COMPR, lane);

  float m_run[8], l_run[8];
  v8f o[4] = {};
#pragma unroll
  for (int r = 0; r < 8; ++r) { m_run[r] = -INFINITY; l_run[r] = 0.0f; }

  const int half = lane >> 4;
  const int nl   = lane & 15;
  const float scale = 0.125f;  // 1/sqrt(64)

  const int nchunks = q0 / 32 + 1;  // last k0 <= q0 (causal coverage)
  stage(0, 0);

  for (int kb = 0; kb < nchunks; ++kb) {
    const int k0 = kb * 32;
    const int buf = kb & 1;

    if (kb + 1 < nchunks) {
      stage(k0 + 32, buf ^ 1);  // prefetch next chunk
      WAIT_STAGE_PREFETCH();    // current chunk's DMA has completed
    } else {
      WAIT_STAGE_ALL();
    }
    const bf16* Kt = &Klds[buf][0];
    const bf16* Vt = &Vlds[buf][0];

    // ---- scores: two 16x16 tiles over k-positions (from LDS) ----
    v8f c0 = {}, c1 = {};
    {
      v16bf bK;
      bK = load_bT_frag(Kt + 0, 64, lane);
      c0 = wmma_bf16(aQ0, bK, c0);
      bK = load_bT_frag(Kt + 32, 64, lane);
      c0 = wmma_bf16(aQ1, bK, c0);
      bK = load_bT_frag(Kt + 16 * 64 + 0, 64, lane);
      c1 = wmma_bf16(aQ0, bK, c1);
      bK = load_bT_frag(Kt + 16 * 64 + 32, 64, lane);
      c1 = wmma_bf16(aQ1, bK, c1);
    }

    // ---- scale + causal mask ----
    float s0[8], s1[8];
    const bool needmask = (k0 + 31 > q0);
#pragma unroll
    for (int r = 0; r < 8; ++r) {
      const int rowg = q0 + r + half * 8;
      s0[r] = c0[r] * scale;
      s1[r] = c1[r] * scale;
      if (needmask) {
        if (k0 + nl > rowg)      s0[r] = -INFINITY;
        if (k0 + 16 + nl > rowg) s1[r] = -INFINITY;
      }
    }

    // ---- row max across the 16 lanes of this half ----
    float mx[8];
#pragma unroll
    for (int r = 0; r < 8; ++r) mx[r] = fmaxf(s0[r], s1[r]);
#pragma unroll
    for (int off = 1; off < 16; off <<= 1) {
#pragma unroll
      for (int r = 0; r < 8; ++r)
        mx[r] = fmaxf(mx[r], __shfl_xor(mx[r], off, 32));
    }

    // ---- online softmax update ----
    float p0[8], p1[8], rs[8], alpha[8];
#pragma unroll
    for (int r = 0; r < 8; ++r) {
      const float mn = fmaxf(m_run[r], mx[r]);  // finite: col k0 always valid
      alpha[r] = __expf(m_run[r] - mn);         // first iter: exp(-inf) = 0
      p0[r] = __expf(s0[r] - mn);
      p1[r] = __expf(s1[r] - mn);
      rs[r] = p0[r] + p1[r];
      m_run[r] = mn;
    }
#pragma unroll
    for (int off = 1; off < 16; off <<= 1) {
#pragma unroll
      for (int r = 0; r < 8; ++r) rs[r] += __shfl_xor(rs[r], off, 32);
    }
#pragma unroll
    for (int r = 0; r < 8; ++r) l_run[r] = l_run[r] * alpha[r] + rs[r];
#pragma unroll
    for (int t = 0; t < 4; ++t) {
#pragma unroll
      for (int r = 0; r < 8; ++r) o[t][r] *= alpha[r];
    }

    // ---- C-layout P -> LDS (16x32 tile) -> A-fragment ----
#pragma unroll
    for (int r = 0; r < 8; ++r) {
      Plds[(r + half * 8) * 32 + nl]      = (bf16)p0[r];
      Plds[(r + half * 8) * 32 + 16 + nl] = (bf16)p1[r];
    }
    __syncthreads();
    v16bf aP = load_a_frag(Plds, 32, lane);

    // ---- O += P * V over four 16-wide d-tiles (V from LDS) ----
#pragma unroll
    for (int t = 0; t < 4; ++t) {
      v16bf bV = load_b_frag(Vt + t * 16, 64, lane);
      o[t] = wmma_bf16(aP, bV, o[t]);
    }
    __syncthreads();
  }

  // ---- normalize and store ----
#pragma unroll
  for (int t = 0; t < 4; ++t) {
#pragma unroll
    for (int r = 0; r < 8; ++r) {
      const size_t rowg = (size_t)b * SEQ + q0 + r + half * 8;
      const float v = o[t][r] / l_run[r];
      O[rowg * D_COMPR + h * HDIM + t * 16 + nl] = (bf16)v;
    }
  }
}

// ---------------------------------------------------------------------------
// Host-side orchestration
// ---------------------------------------------------------------------------
static inline void* ws_alloc(void* base, size_t& off, size_t bytes) {
  void* p = (char*)base + off;
  off += (bytes + 255) & ~(size_t)255;
  return p;
}

extern "C" void kernel_launch(void* const* d_in, const int* in_sizes, int n_in,
                              void* d_out, int out_size, void* d_ws,
                              size_t ws_size, hipStream_t stream) {
  (void)in_sizes; (void)n_in; (void)out_size; (void)ws_size;

  const float* x     = (const float*)d_in[0];
  const float* CQ_w  = (const float*)d_in[1];
  const float* CQ_b  = (const float*)d_in[2];
  const float* CQC_w = (const float*)d_in[3];
  const float* CQC_b = (const float*)d_in[4];
  const float* CQR_w = (const float*)d_in[5];
  const float* CQR_b = (const float*)d_in[6];
  const float* CKV_w = (const float*)d_in[7];
  const float* CKV_b = (const float*)d_in[8];
  const float* CKR_w = (const float*)d_in[9];
  const float* CKR_b = (const float*)d_in[10];
  const float* CKC_w = (const float*)d_in[11];
  const float* CKC_b = (const float*)d_in[12];
  const float* CV_w  = (const float*)d_in[13];
  const float* CV_b  = (const float*)d_in[14];
  const float* OUT_w = (const float*)d_in[15];
  const float* OUT_b = (const float*)d_in[16];

  size_t off = 0;
  bf16* xb   = (bf16*)ws_alloc(d_ws, off, (size_t)NTOK * EMBED * 2);
  bf16* wq   = (bf16*)ws_alloc(d_ws, off, (size_t)D_COMPR * EMBED * 2);
  bf16* wqc  = (bf16*)ws_alloc(d_ws, off, (size_t)(D_COMPR - D_ROPE) * D_COMPR * 2);
  bf16* wqr  = (bf16*)ws_alloc(d_ws, off, (size_t)D_ROPE * EMBED * 2);
  bf16* wkv  = (bf16*)ws_alloc(d_ws, off, (size_t)D_COMPR * EMBED * 2);
  bf16* wkr  = (bf16*)ws_alloc(d_ws, off, (size_t)D_ROPE * EMBED * 2);
  bf16* wkc  = (bf16*)ws_alloc(d_ws, off, (size_t)(D_COMPR - D_ROPE) * D_COMPR * 2);
  bf16* wv   = (bf16*)ws_alloc(d_ws, off, (size_t)D_COMPR * D_COMPR * 2);
  bf16* wout = (bf16*)ws_alloc(d_ws, off, (size_t)EMBED * D_COMPR * 2);
  bf16* cq   = (bf16*)ws_alloc(d_ws, off, (size_t)NTOK * D_COMPR * 2);
  bf16* ckv  = (bf16*)ws_alloc(d_ws, off, (size_t)NTOK * D_COMPR * 2);
  bf16* Qb   = (bf16*)ws_alloc(d_ws, off, (size_t)NTOK * D_COMPR * 2);
  bf16* Kb   = (bf16*)ws_alloc(d_ws, off, (size_t)NTOK * D_COMPR * 2);
  bf16* Vb   = (bf16*)ws_alloc(d_ws, off, (size_t)NTOK * D_COMPR * 2);
  float* cqr = (float*)ws_alloc(d_ws, off, (size_t)NTOK * D_ROPE * 4);
  float* ckr = (float*)ws_alloc(d_ws, off, (size_t)NTOK * D_ROPE * 4);
  bf16* att  = (bf16*)ws_alloc(d_ws, off, (size_t)NTOK * D_COMPR * 2);

  auto cvt = [&](const float* s, bf16* d, long n) {
    f32_to_bf16_kernel<<<(int)((n + 255) / 256), 256, 0, stream>>>(s, d, n);
  };
  cvt(x,     xb,   (long)NTOK * EMBED);
  cvt(CQ_w,  wq,   (long)D_COMPR * EMBED);
  cvt(CQC_w, wqc,  (long)(D_COMPR - D_ROPE) * D_COMPR);
  cvt(CQR_w, wqr,  (long)D_ROPE * EMBED);
  cvt(CKV_w, wkv,  (long)D_COMPR * EMBED);
  cvt(CKR_w, wkr,  (long)D_ROPE * EMBED);
  cvt(CKC_w, wkc,  (long)(D_COMPR - D_ROPE) * D_COMPR);
  cvt(CV_w,  wv,   (long)D_COMPR * D_COMPR);
  cvt(OUT_w, wout, (long)EMBED * D_COMPR);

  auto gemm = [&](const bf16* A, int lda, const bf16* W, int ldw,
                  const float* bias, bf16* ob, float* of, int ldc, int coloff,
                  int N, int K, int gelu) {
    dim3 grid(NTOK / 128, N / 32);
    gemm_bf16_wmma<<<grid, 128, 0, stream>>>(A, lda, W, ldw, bias, ob, of,
                                             ldc, coloff, K, gelu);
  };

  // projections from x
  gemm(xb, EMBED, wq,  EMBED, CQ_b,  cq,  nullptr, D_COMPR, 0, D_COMPR, EMBED, 1);
  gemm(xb, EMBED, wkv, EMBED, CKV_b, ckv, nullptr, D_COMPR, 0, D_COMPR, EMBED, 1);
  gemm(xb, EMBED, wqr, EMBED, CQR_b, nullptr, cqr, D_ROPE, 0, D_ROPE, EMBED, 0);
  gemm(xb, EMBED, wkr, EMBED, CKR_b, nullptr, ckr, D_ROPE, 0, D_ROPE, EMBED, 0);

  // second-level projections; gelu is elementwise so gelu(concat) splits
  gemm(cq,  D_COMPR, wqc, D_COMPR, CQC_b, Qb, nullptr, D_COMPR, 0,
       D_COMPR - D_ROPE, D_COMPR, 1);
  gemm(ckv, D_COMPR, wkc, D_COMPR, CKC_b, Kb, nullptr, D_COMPR, 0,
       D_COMPR - D_ROPE, D_COMPR, 1);
  gemm(ckv, D_COMPR, wv,  D_COMPR, CV_b,  Vb, nullptr, D_COMPR, 0,
       D_COMPR, D_COMPR, 0);

  // rope + gelu into the last 128 columns of Q/K
  {
    const int n = NTOK * 64;
    rope_gelu_kernel<<<(n + 255) / 256, 256, 0, stream>>>(cqr, Qb);
    rope_gelu_kernel<<<(n + 255) / 256, 256, 0, stream>>>(ckr, Kb);
  }

  // causal flash attention
  {
    dim3 grid(SEQ / 16, NHEADS, BATCH);
    mla_attention_kernel<<<grid, 32, 0, stream>>>(Qb, Kb, Vb, att);
  }

  // output projection -> fp32 result
  gemm(att, D_COMPR, wout, D_COMPR, OUT_b, nullptr, (float*)d_out, EMBED, 0,
       EMBED, D_COMPR, 0);
}
